// SeqQuery_6511170421698
// MI455X (gfx1250) — matrix operand order repaired and checked by reading, not compiled, not run on hardware
//
#include <hip/hip_runtime.h>
#include <hip/hip_bf16.h>

typedef __attribute__((ext_vector_type(16))) _Float16 v16h;
typedef __attribute__((ext_vector_type(8)))  float    v8f;

#define HDIM 128   // hidden size (fixed by the reference)
#define TILE 16    // rows per wave tile
#define WPB  8     // waves per block (256 threads, wave32)

// ---------------- prep kernels ----------------
__global__ void cvt_f32_to_f16(const float* __restrict__ src,
                               _Float16* __restrict__ dst, int n) {
    int i = blockIdx.x * blockDim.x + threadIdx.x;
    if (i < n) dst[i] = (_Float16)src[i];
}

__global__ void zero_f32(float* __restrict__ p, int n) {
    int i = blockIdx.x * blockDim.x + threadIdx.x;
    if (i < n) p[i] = 0.0f;
}

// ---------------- T = query @ W1^T + W1_b + W2_b  (B x H) ----------------
// one block per segment row; query row staged through LDS
__global__ __launch_bounds__(HDIM) void seg_T_kernel(
    const float* __restrict__ query, const float* __restrict__ W1,
    const float* __restrict__ W1b, const float* __restrict__ W2b,
    float* __restrict__ T) {
    __shared__ float q[HDIM];
    const int b = blockIdx.x;
    const int h = threadIdx.x;
    q[h] = query[(size_t)b * HDIM + h];
    __syncthreads();
    const float* wrow = W1 + (size_t)h * HDIM;
    float s = W1b[h] + W2b[h];
#pragma unroll 8
    for (int k = 0; k < HDIM; ++k) s += q[k] * wrow[k];
    T[(size_t)b * HDIM + h] = s;
}

// ---------------- main fused kernel ----------------
// per wave: 16-row tile of sess_embed
//   S = sess_tile @ W2^T            (32x v_wmma_f32_16x16x32_f16)
//   h = sigmoid(S + T[seg])         (T gathered per row)
//   w = h . alpha + alpha_b         (butterfly shfl_xor reduce, wave32)
//   out[seg] += w * sess_row        (run-length local acc, atomic flush)
__global__ __launch_bounds__(256) void seg_main_kernel(
    const float* __restrict__ sess, const int* __restrict__ segids,
    const _Float16* __restrict__ W2h, const float* __restrict__ T,
    const float* __restrict__ alpha, const float* __restrict__ alphab,
    float* __restrict__ out, long N) {
    const int lane = threadIdx.x & 31;
    const long tile = (long)blockIdx.x * WPB + (threadIdx.x >> 5);
    const long base = tile * TILE;
    if (base >= N) return;            // uniform per wave

    const int lrow = lane & 15;       // row (A) / col (B,C) within 16
    const int lhi  = lane >> 4;       // half-wave select

    // segment ids for the 8 rows this half-wave owns (rows lhi*8 + j)
    int sj[8];
#pragma unroll
    for (int j = 0; j < 8; ++j) sj[j] = segids[base + lhi * 8 + j];

    v8f c[8] = {};                    // 8 N-tiles of 16x16 f32 accumulators

    const float* arow = sess + (base + lrow) * (long)HDIM;
#pragma unroll
    for (int kk = 0; kk < 4; ++kk) {  // K steps of 32
        // A layout (16-bit 16x32): lanes 0-15 rows, halves 0-15 = K[kb..kb+15],
        // kb = lhi*16 (contiguous, matches B)
        const int k0 = kk * 32 + lhi * 16;
        const int k1 = k0 + 8;
        float fa[16];
        *(float4*)(fa + 0)  = ((const float4*)(arow + k0))[0];
        *(float4*)(fa + 4)  = ((const float4*)(arow + k0))[1];
        *(float4*)(fa + 8)  = ((const float4*)(arow + k1))[0];
        *(float4*)(fa + 12) = ((const float4*)(arow + k1))[1];
        v16h a;
#pragma unroll
        for (int i = 0; i < 16; ++i) a[i] = (_Float16)fa[i];

#pragma unroll
        for (int t = 0; t < 8; ++t) { // N tiles
            // B layout (32x16): lane = column n, halves = 16 contiguous K
            const int n = t * 16 + lrow;
            const v16h b = *(const v16h*)(W2h + (size_t)n * HDIM + kk * 32 + lhi * 16);
            c[t] = __builtin_amdgcn_wmma_f32_16x16x32_f16(
                false, a, false, b, (short)0, c[t], false, false);
        }
    }

    // h = sigmoid(S + T[seg]); partial dot with alpha
    float wj[8];
#pragma unroll
    for (int j = 0; j < 8; ++j) wj[j] = 0.0f;
#pragma unroll
    for (int t = 0; t < 8; ++t) {
        const int ncol = t * 16 + lrow;
        const float av = alpha[ncol];
#pragma unroll
        for (int j = 0; j < 8; ++j) {
            const float x = c[t][j] + T[(long)sj[j] * HDIM + ncol];
            const float hsig = 1.0f / (1.0f + __expf(-x));
            wj[j] += hsig * av;
        }
    }
    const float ab = alphab[0];
#pragma unroll
    for (int j = 0; j < 8; ++j) {
#pragma unroll
        for (int m = 1; m < 16; m <<= 1)        // reduce within 16-lane half
            wj[j] += __shfl_xor(wj[j], m, 32);
        wj[j] += ab;
    }

    // segmented accumulation: lane owns 4 columns; run-length over sorted ids
    const int colb = lane * 4;
    float4 acc = make_float4(0.f, 0.f, 0.f, 0.f);
    int cur = -1;
#pragma unroll
    for (int m = 0; m < TILE; ++m) {
        const int src = (m < 8) ? 0 : 16;
        const int   s = __shfl(sj[m & 7], src, 32);
        const float w = __shfl(wj[m & 7], src, 32);
        if (s != cur) {
            if (cur >= 0) {
                float* o = out + (long)cur * HDIM + colb;
                atomicAdd(o + 0, acc.x); atomicAdd(o + 1, acc.y);
                atomicAdd(o + 2, acc.z); atomicAdd(o + 3, acc.w);
            }
            acc = make_float4(0.f, 0.f, 0.f, 0.f);
            cur = s;
        }
        const float4 x = *(const float4*)(sess + (base + m) * (long)HDIM + colb);
        acc.x += w * x.x; acc.y += w * x.y; acc.z += w * x.z; acc.w += w * x.w;
    }
    if (cur >= 0) {
        float* o = out + (long)cur * HDIM + colb;
        atomicAdd(o + 0, acc.x); atomicAdd(o + 1, acc.y);
        atomicAdd(o + 2, acc.z); atomicAdd(o + 3, acc.w);
    }
}

extern "C" void kernel_launch(void* const* d_in, const int* in_sizes, int n_in,
                              void* d_out, int out_size, void* d_ws, size_t ws_size,
                              hipStream_t stream) {
    const float* sess   = (const float*)d_in[0];
    const float* query  = (const float*)d_in[1];
    const int*   segids = (const int*)d_in[2];
    // d_in[3] = num_segments (device scalar, derived from out_size instead)
    const float* W1     = (const float*)d_in[4];
    const float* W1b    = (const float*)d_in[5];
    const float* W2     = (const float*)d_in[6];
    const float* W2b    = (const float*)d_in[7];
    const float* alpha  = (const float*)d_in[8];
    const float* alphab = (const float*)d_in[9];
    float* out = (float*)d_out;

    const int  H = in_sizes[5];            // W1_b size
    const long N = (long)in_sizes[0] / H;  // sess rows
    const int  B = out_size / H;           // segments

    // workspace: [W2 as f16 | T (BxH f32)]
    _Float16* W2h = (_Float16*)d_ws;
    size_t toff = ((size_t)H * H * sizeof(_Float16) + 255) & ~(size_t)255;
    float* T = (float*)((char*)d_ws + toff);

    cvt_f32_to_f16<<<(H * H + 255) / 256, 256, 0, stream>>>(W2, W2h, H * H);
    zero_f32<<<(out_size + 255) / 256, 256, 0, stream>>>(out, out_size);
    seg_T_kernel<<<B, HDIM, 0, stream>>>(query, W1, W1b, W2b, T);

    const long tiles  = (N + TILE - 1) / TILE;
    const long blocks = (tiles + WPB - 1) / WPB;
    seg_main_kernel<<<(int)blocks, 32 * WPB, 0, stream>>>(
        sess, segids, W2h, T, alpha, alphab, out, N);
}
